// SimpleEdgeModel_18683107737901
// MI455X (gfx1250) — matrix operand order repaired and checked by reading, not compiled, not run on hardware
//
#include <hip/hip_runtime.h>

// ---------------------------------------------------------------------------
// SimpleEdgeModel for MI455X (gfx1250).
// Dominant cost: two 512x512 GEMMs over 294,912 edge rows (~310 GFLOP) ->
// fused WMMA (bf16 in, f32 accum) edge kernel with M-register-blocking so
// each B (weight) fragment fetched from L2 feeds 4 WMMAs.
// ---------------------------------------------------------------------------

typedef __attribute__((ext_vector_type(16))) __bf16 v16bf;
typedef __attribute__((ext_vector_type(8)))  __bf16 v8bf;
typedef __attribute__((ext_vector_type(8)))  float  v8f;

static constexpr int Bq = 8;
static constexpr int Nq = 192;
static constexpr int Dq = 128;
static constexpr int Hq = 512;
static constexpr int JT = 64;             // j-rows per edge block
static constexpr int ROWS = Bq * Nq;      // 1536 token rows
static constexpr int EMBK = 5 * Dq;       // 640

// ---------------------------------------------------------------------------
// Embedding gather (tables with row 0 forced to zero) -> emb (1536 x 640) f32
// ---------------------------------------------------------------------------
__global__ void embed_kernel(const int* __restrict__ bricks,
                             const float* __restrict__ eb,
                             const float* __restrict__ ex,
                             const float* __restrict__ ey,
                             const float* __restrict__ ez,
                             float* __restrict__ emb)
{
    int idx = blockIdx.x * blockDim.x + threadIdx.x;   // over 1536*640
    if (idx >= ROWS * EMBK) return;
    int r = idx / EMBK, k = idx % EMBK;
    int b = r / Nq, n = r % Nq;
    int s = k / Dq, d = k % Dq;
    int bi = bricks[(b * 5 + s) * Nq + n];
    const float* tab = (s == 0) ? eb : (s == 2) ? ey : (s == 3) ? ez : ex;
    emb[idx] = (bi == 0) ? 0.0f : tab[bi * Dq + d];
}

// ---------------------------------------------------------------------------
// Generic small fp32 GEMM: C[r][h] = (relu)(sum_k A[r][k] * W[k][h] + bias[h])
// Used only for the tiny token-level layers (~1% of total FLOPs).
// ---------------------------------------------------------------------------
__global__ void gemm_f32(const float* __restrict__ A,
                         const float* __restrict__ W,
                         const float* __restrict__ bias,
                         float* __restrict__ C,
                         int M, int K, int Nn, int do_relu)
{
    int idx = blockIdx.x * blockDim.x + threadIdx.x;
    if (idx >= M * Nn) return;
    int r = idx / Nn, h = idx % Nn;
    float acc = bias ? bias[h] : 0.0f;
    const float* a = A + (size_t)r * K;
    for (int k = 0; k < K; ++k) acc = fmaf(a[k], W[(size_t)k * Nn + h], acc);
    if (do_relu) acc = acc > 0.0f ? acc : 0.0f;
    C[idx] = acc;
}

// ---------------------------------------------------------------------------
// Swizzle a 512x512 row-major fp32 weight into per-fragment bf16 layout for
// the WMMA B operand (B is 32x16 per fragment).  B layout (contiguous k, same as
// this file's A loads, not AMD's table order): lane l holds column n = l%16; lanes 0-15 hold
// K = kt*32 + 0..15, lanes 16-31 hold K = kt*32 + 16..31, 16 contiguous bf16
// per lane.  Fragment f = kt*32 + nt stored as 32 lanes x 16 bf16.
// ---------------------------------------------------------------------------
__global__ void swizzle_w(const float* __restrict__ W, __bf16* __restrict__ Wsw)
{
    int idx = blockIdx.x * blockDim.x + threadIdx.x;   // over 512*512
    if (idx >= Hq * Hq) return;
    int e = idx & 15;
    int l = (idx >> 4) & 31;
    int f = idx >> 9;
    int kt = f >> 5, nt = f & 31;
    int k = kt * 32 + (l >> 4) * 16 + e;
    int n = nt * 16 + (l & 15);
    Wsw[idx] = (__bf16)W[(size_t)k * Hq + n];
}

// ---------------------------------------------------------------------------
// One WMMA GEMM stage: dst = relu(src @ Wsw + bias), 64x512 bf16 tiles.
// Each wave owns ALL 4 m-tiles and 4 n-tiles [ntB, ntB+4).  Inner step loads
// 4 A fragments + 2 B fragments and issues 8 WMMAs: each L2 B-fragment feeds
// 4 WMMAs (4x less L2 traffic), each LDS A-fragment feeds 2 WMMAs.
// Every weight fragment is read exactly once per block per stage.
// ---------------------------------------------------------------------------
__device__ __forceinline__ void wmma_stage(const __bf16* __restrict__ src,
                                           __bf16* __restrict__ dst,
                                           const __bf16* __restrict__ Wsw,
                                           const float* __restrict__ bias,
                                           int ntB, int lrow, int lhalf)
{
    const int lane = (threadIdx.x & 31);
    for (int p = 0; p < 2; ++p) {
        const int nt0 = ntB + 2 * p;
        v8f acc0[4] = {{}, {}, {}, {}};
        v8f acc1[4] = {{}, {}, {}, {}};
        #pragma unroll 2
        for (int kt = 0; kt < 16; ++kt) {
            const int kb = kt * 32 + lhalf * 16;
            union { v16bf v; v8bf h[2]; } a[4], b0, b1;
            const __bf16* wp = &Wsw[(size_t)((kt * 32 + nt0) * 32 + lane) * 16];
            b0.v = *(const v16bf*)wp;
            b1.v = *(const v16bf*)(wp + 512);   // next n-tile fragment
            #pragma unroll
            for (int m = 0; m < 4; ++m) {
                const __bf16* Arow = &src[(m * 16 + lrow) * Hq + kb];
                a[m].h[0] = *(const v8bf*)Arow;
                a[m].h[1] = *(const v8bf*)(Arow + 8);
            }
            #pragma unroll
            for (int m = 0; m < 4; ++m) {
                acc0[m] = __builtin_amdgcn_wmma_f32_16x16x32_bf16(
                            false, a[m].v, false, b0.v, (short)0, acc0[m], false, false);
                acc1[m] = __builtin_amdgcn_wmma_f32_16x16x32_bf16(
                            false, a[m].v, false, b1.v, (short)0, acc1[m], false, false);
            }
        }
        // epilogue: bias + relu + bf16 store to LDS (C layout: VGPR e holds
        // row m*16 + lhalf*8 + e, column nt*16 + lrow)
        const int col0 = nt0 * 16 + lrow;
        const float bv0 = bias[col0];
        const float bv1 = bias[col0 + 16];
        #pragma unroll
        for (int m = 0; m < 4; ++m) {
            #pragma unroll
            for (int e = 0; e < 8; ++e) {
                float r0 = acc0[m][e] + bv0;
                float r1 = acc1[m][e] + bv1;
                __bf16* drow = &dst[(m * 16 + lhalf * 8 + e) * Hq];
                drow[col0]      = (__bf16)(r0 > 0.0f ? r0 : 0.0f);
                drow[col0 + 16] = (__bf16)(r1 > 0.0f ? r1 : 0.0f);
            }
        }
    }
}

// ---------------------------------------------------------------------------
// Fused edge kernel. One block = (b, i, 64 j's). 256 threads = 8 waves.
//   X  = relu(pj[b,j,:] + pi[b,i,:] + b_ca)          -> ldsA (bf16)
//   H1 = relu(X  @ Wcb + b_cb)  [WMMA bf16]          -> ldsB (bf16)
//   H2 = relu(H1 @ Wcc + b_cc)  [WMMA bf16]          -> ldsA (bf16)
//   out= H2 @ W_out + b_out                          -> global f32
// ---------------------------------------------------------------------------
__global__ __launch_bounds__(256) void edge_kernel(
    const float* __restrict__ pj, const float* __restrict__ pi,
    const float* __restrict__ b_ca, const float* __restrict__ b_cb,
    const float* __restrict__ b_cc, const float* __restrict__ W_out,
    const float* __restrict__ b_out,
    const __bf16* __restrict__ WswB, const __bf16* __restrict__ WswC,
    float* __restrict__ out)
{
    __shared__ __attribute__((aligned(16))) __bf16 ldsA[JT * Hq];  // 64 KB
    __shared__ __attribute__((aligned(16))) __bf16 ldsB[JT * Hq];  // 64 KB
    __shared__ __attribute__((aligned(16))) float  ldsPi[Hq];      //  2 KB

    const int blk = blockIdx.x;
    const int jt  = blk % 3;
    const int bi  = blk / 3;            // b*192 + i
    const int b   = bi / Nq;
    const int j0  = jt * JT;
    const int tid = threadIdx.x;

    // pi row -> LDS
    const float* piRow = pi + (size_t)bi * Hq;
    for (int k = tid; k < Hq; k += 256) ldsPi[k] = piRow[k];
    __syncthreads();

    // Build X in bf16
    const float* pjBase = pj + ((size_t)b * Nq + j0) * Hq;
    for (int idx = tid; idx < JT * Hq; idx += 256) {
        int m = idx >> 9, k = idx & (Hq - 1);
        float v = pjBase[(size_t)m * Hq + k] + ldsPi[k] + b_ca[k];
        ldsA[idx] = (__bf16)(v > 0.0f ? v : 0.0f);
    }
    __syncthreads();

    const int w     = tid >> 5, lane = tid & 31;
    const int ntB   = w * 4;            // 8 waves x 4 n-tiles = 32 n-tiles
    const int lrow  = lane & 15;
    const int lhalf = lane >> 4;

    wmma_stage(ldsA, ldsB, WswB, b_cb, ntB, lrow, lhalf);   // GEMM1
    __syncthreads();
    wmma_stage(ldsB, ldsA, WswC, b_cc, ntB, lrow, lhalf);   // GEMM2
    __syncthreads();

    // Output head: 512 -> 2 per edge row, vectorized v8bf LDS reads
    if (tid < 2 * JT) {
        const int row = tid >> 1, col = tid & 1;
        const v8bf* h2 = (const v8bf*)&ldsA[row * Hq];
        float s = 0.0f;
        for (int k8 = 0; k8 < Hq / 8; ++k8) {
            v8bf hv = h2[k8];
            #pragma unroll
            for (int e = 0; e < 8; ++e)
                s = fmaf((float)hv[e], W_out[(k8 * 8 + e) * 2 + col], s);
        }
        out[(((size_t)bi) * Nq + (j0 + row)) * 2 + col] = s + b_out[col];
    }
}

// ---------------------------------------------------------------------------
// Host-side launch sequence (all on `stream`; graph-capture safe).
// ---------------------------------------------------------------------------
extern "C" void kernel_launch(void* const* d_in, const int* in_sizes, int n_in,
                              void* d_out, int out_size, void* d_ws, size_t ws_size,
                              hipStream_t stream)
{
    (void)in_sizes; (void)n_in; (void)out_size; (void)ws_size;

    const int*   bricks = (const int*)  d_in[0];
    const float* e_brick= (const float*)d_in[1];
    const float* e_x    = (const float*)d_in[2];
    const float* e_y    = (const float*)d_in[3];
    const float* e_z    = (const float*)d_in[4];
    const float* W_a    = (const float*)d_in[5];
    const float* b_a    = (const float*)d_in[6];
    const float* W_b    = (const float*)d_in[7];
    const float* b_b    = (const float*)d_in[8];
    const float* W_ca   = (const float*)d_in[9];
    const float* b_ca   = (const float*)d_in[10];
    const float* W_cb   = (const float*)d_in[11];
    const float* b_cb   = (const float*)d_in[12];
    const float* W_cc   = (const float*)d_in[13];
    const float* b_cc   = (const float*)d_in[14];
    const float* W_out  = (const float*)d_in[15];
    const float* b_out  = (const float*)d_in[16];
    float* out = (float*)d_out;

    // Workspace layout (bytes). pj/pi reuse emb/feats_a regions (dead by then).
    char* ws = (char*)d_ws;
    const size_t embBytes = (size_t)ROWS * EMBK * sizeof(float);   // 3,932,160
    const size_t rowBytes = (size_t)ROWS * Hq   * sizeof(float);   // 3,145,728
    float*  emb     = (float*)(ws);
    float*  feats_a = (float*)(ws + embBytes);
    float*  feats_b = (float*)(ws + embBytes + rowBytes);
    float*  pjBuf   = (float*)(ws);                 // reuse emb region
    float*  piBuf   = (float*)(ws + embBytes);      // reuse feats_a region
    __bf16* WswB    = (__bf16*)(ws + embBytes + 2 * rowBytes);
    __bf16* WswC    = (__bf16*)(ws + embBytes + 2 * rowBytes
                                + (size_t)Hq * Hq * sizeof(__bf16));

    const int TB = 256;

    // 1) Embedding gather
    embed_kernel<<<(ROWS * EMBK + TB - 1) / TB, TB, 0, stream>>>(
        bricks, e_brick, e_x, e_y, e_z, emb);

    // 2) feats = relu(emb @ W_a + b_a)
    gemm_f32<<<(ROWS * Hq + TB - 1) / TB, TB, 0, stream>>>(
        emb, W_a, b_a, feats_a, ROWS, EMBK, Hq, 1);

    // 3) feats = relu(feats @ W_b + b_b)
    gemm_f32<<<(ROWS * Hq + TB - 1) / TB, TB, 0, stream>>>(
        feats_a, W_b, b_b, feats_b, ROWS, Hq, Hq, 1);

    // 4) pj = feats @ W_ca[:H],  pi = feats @ W_ca[H:]
    gemm_f32<<<(ROWS * Hq + TB - 1) / TB, TB, 0, stream>>>(
        feats_b, W_ca, nullptr, pjBuf, ROWS, Hq, Hq, 0);
    gemm_f32<<<(ROWS * Hq + TB - 1) / TB, TB, 0, stream>>>(
        feats_b, W_ca + (size_t)Hq * Hq, nullptr, piBuf, ROWS, Hq, Hq, 0);

    // 5) Weight swizzle to WMMA-B bf16 fragments
    swizzle_w<<<(Hq * Hq + TB - 1) / TB, TB, 0, stream>>>(W_cb, WswB);
    swizzle_w<<<(Hq * Hq + TB - 1) / TB, TB, 0, stream>>>(W_cc, WswC);

    // 6) Fused edge pipeline (WMMA): 8*192*3 = 4608 blocks
    edge_kernel<<<Bq * Nq * (Nq / JT), TB, 0, stream>>>(
        pjBuf, piBuf, b_ca, b_cb, b_cc, W_out, b_out, WswB, WswC, out);
}
